// SparseMoE_56324201119927
// MI455X (gfx1250) — matrix operand order repaired and checked by reading, not compiled, not run on hardware
//
#include <hip/hip_runtime.h>
#include <hip/hip_bf16.h>

#define D_DIM 768
#define H_DIM 3072
#define E_NUM 8
#define T_NUM 2048
#define TILE_M 32
#define HB 64

// tiled-weight geometry: tile = 32(K) x 16(N) bf16 = 256 dwords, dword = lane*8+v
#define W1_NT (H_DIM / 16)   // 192
#define W1_KT (D_DIM / 32)   // 24
#define W2_NT (D_DIM / 16)   // 48
#define W2_KT (H_DIM / 32)   // 96
#define W1T_DW ((size_t)W1_NT * W1_KT * 256)   // per matrix
#define W2T_DW ((size_t)W2_NT * W2_KT * 256)

typedef __attribute__((ext_vector_type(16))) __bf16 v16bf;
typedef __attribute__((ext_vector_type(8)))  float  v8f;

union B128 { uint4 q[2]; unsigned int u[8]; v16bf v; };

__device__ __forceinline__ unsigned int bf1(float f) {
  unsigned int u = __float_as_uint(f);
  return (u + 0x7FFFu + ((u >> 16) & 1u)) >> 16;  // RNE fp32->bf16
}
__device__ __forceinline__ unsigned int pack2(float f0, float f1) {
  return bf1(f0) | (bf1(f1) << 16);
}
__device__ __forceinline__ v8f wmma_bf16(const B128& A, const B128& B, v8f C) {
  return __builtin_amdgcn_wmma_f32_16x16x32_bf16(false, A.v, false, B.v,
                                                 (short)0, C, false, false);
}

// ------------- Pack fp32 [K,N] row-major -> bf16 WMMA-B tiled layout ---------
// dst[((ntile*(K/32)+kt)*256) + lane*8 + v] packs elems (k,n),(k+1,n)
// with lane = (n&15) + ((klocal&8) ? 16 : 0), v = ((klocal&16)>>2) + ((klocal&7)>>1)
__global__ void pack_w(const float* __restrict__ src, unsigned int* __restrict__ dst,
                       int K, int N) {
  size_t matSrc = (size_t)blockIdx.y * K * N;
  size_t matDst = (size_t)blockIdx.y * (((size_t)(N >> 4) * (K >> 5)) << 8);
  int flat = blockIdx.x * 256 + threadIdx.x;        // (K/2)*N threads
  int n  = flat % N;
  int k  = (flat / N) * 2;
  float f0 = src[matSrc + (size_t)k * N + n];       // coalesced over n
  float f1 = src[matSrc + (size_t)(k + 1) * N + n];
  int klocal = k & 31, kt = k >> 5, ntile = n >> 4;
  int lane = (n & 15) + ((klocal & 8) ? 16 : 0);
  int v = ((klocal & 16) >> 2) + ((klocal & 7) >> 1);
  dst[matDst + (((size_t)(ntile * (K >> 5) + kt)) << 8) + lane * 8 + v] = pack2(f0, f1);
}

// ---------------- Router: noisy logits -> top-2 -> softmax -> gather lists ----
__global__ void moe_router(const float* __restrict__ x,
                           const float* __restrict__ noise,
                           const float* __restrict__ Wr,
                           const float* __restrict__ br,
                           int* __restrict__ cnt,
                           int* __restrict__ listSlot,
                           float* __restrict__ listGate) {
  int wave = (blockIdx.x * blockDim.x + threadIdx.x) >> 5;  // one wave = one token
  int lane = threadIdx.x & 31;
  if (wave >= T_NUM) return;
  int t = wave;

  float acc[E_NUM];
#pragma unroll
  for (int e = 0; e < E_NUM; ++e) acc[e] = 0.f;
  for (int d = lane; d < D_DIM; d += 32) {
    float xv = x[t * D_DIM + d];
#pragma unroll
    for (int e = 0; e < E_NUM; ++e) acc[e] += xv * Wr[d * E_NUM + e];
  }
#pragma unroll
  for (int e = 0; e < E_NUM; ++e) {
#pragma unroll
    for (int off = 16; off > 0; off >>= 1) acc[e] += __shfl_xor(acc[e], off, 32);
  }
  if (lane == 0) {
#pragma unroll
    for (int e = 0; e < E_NUM; ++e) acc[e] += br[e] + 0.1f * noise[t * E_NUM + e];
    int i0 = 0; float v0 = acc[0];
    for (int e = 1; e < E_NUM; ++e) if (acc[e] > v0) { v0 = acc[e]; i0 = e; }
    int i1 = -1; float v1 = -3.4e38f;
    for (int e = 0; e < E_NUM; ++e) {
      if (e == i0) continue;
      if (acc[e] > v1) { v1 = acc[e]; i1 = e; }
    }
    float ex = __expf(v1 - v0);
    float g0 = 1.f / (1.f + ex);
    float g1 = ex / (1.f + ex);
    int p0 = atomicAdd(&cnt[i0], 1);
    listSlot[i0 * T_NUM + p0] = 2 * t;     listGate[i0 * T_NUM + p0] = g0;
    int p1 = atomicAdd(&cnt[i1], 1);
    listSlot[i1 * T_NUM + p1] = 2 * t + 1; listGate[i1 * T_NUM + p1] = g1;
  }
}

// ---------------- Expert / shared FFN over gathered token tiles (WMMA bf16) ---
__global__ __launch_bounds__(256, 1)
void moe_ffn(const float* __restrict__ x,
             const unsigned int* __restrict__ W1t, const float* __restrict__ b1,
             const unsigned int* __restrict__ W2t, const float* __restrict__ b2,
             const unsigned int* __restrict__ Ws1t, const float* __restrict__ bs1,
             const unsigned int* __restrict__ Ws2t, const float* __restrict__ bs2,
             const int* __restrict__ cnt,
             const int* __restrict__ listSlot, const float* __restrict__ listGate,
             float* __restrict__ partial, float* __restrict__ sharedOut) {
  const int TILES = T_NUM / TILE_M;          // 64 tiles per expert (worst case)
  int e    = blockIdx.x / TILES;
  int tile = blockIdx.x % TILES;
  int m0   = tile * TILE_M;
  bool isShared = (e == E_NUM);
  int count = isShared ? T_NUM : cnt[e];
  if (m0 >= count) return;

  const unsigned int* w1t = isShared ? Ws1t : (W1t + (size_t)e * W1T_DW);
  const float*        b1p = isShared ? bs1  : (b1 + e * H_DIM);
  const unsigned int* w2t = isShared ? Ws2t : (W2t + (size_t)e * W2T_DW);
  const float*        b2p = isShared ? bs2  : (b2 + e * D_DIM);

  // x tile pre-swizzled into WMMA A layout: 2 mrow-tiles x 24 k-tiles x 64 uint4
  __shared__ uint4 x_a4[2 * 24 * 64];                 // 48 KB
  __shared__ uint4 h_a4[4 * 64];                      // 4 KB : 2 mrow x 2 ktile
  __shared__ int   slot_l[TILE_M];
  __shared__ float gate_l[TILE_M];
  __shared__ int   tok_l[TILE_M];

  int tid = threadIdx.x;
  if (tid < TILE_M) {
    int sIdx = m0 + tid;
    int slot = -1; float g = 0.f; int tok = 0;
    if (sIdx < count) {
      if (isShared) { slot = sIdx; g = 1.f; tok = sIdx; }
      else {
        slot = listSlot[e * T_NUM + sIdx];
        g    = listGate[e * T_NUM + sIdx];
        tok  = slot >> 1;
      }
    }
    slot_l[tid] = slot; gate_l[tid] = g; tok_l[tid] = tok;
  }
  __syncthreads();

  // gather token rows -> bf16, directly in WMMA-A swizzled order
  {
    unsigned int* xu = (unsigned int*)x_a4;
    for (int idx = tid; idx < 2 * 24 * 256; idx += 256) {
      int tileI = idx >> 8;                  // mrowT*24 + kt
      int mrowT = tileI / 24, kt = tileI - mrowT * 24;
      int dw = idx & 255;
      int lp = dw >> 3, v = dw & 7;
      int m  = lp & 15;
      int koff = (lp >= 16) ? 8 : 0;
      int kL = ((v < 4) ? 2 * v : 16 + 2 * (v - 4)) + koff;
      int row = mrowT * 16 + m;
      const float* xr = x + (size_t)tok_l[row] * D_DIM + kt * 32 + kL;
      xu[idx] = pack2(xr[0], xr[1]);
    }
  }
  __syncthreads();

  int waveId = tid >> 5, lane = tid & 31;
  int mrow  = waveId >> 2;   // 0..1 : 16-row half of the 32-token tile
  int wc    = waveId & 3;    // 0..3 : column group
  int lhalf = lane >> 4;     // 0/1
  int l15   = lane & 15;

  // precomputed h-store swizzle constants (per lane, column cl = wc*16 + l15)
  int cl   = wc * 16 + l15;           // local h column 0..63
  int ktl  = cl >> 5;                 // which h k-tile (0/1)
  int kl   = cl & 31;
  int hp   = kl & 1;
  int kl0  = kl & ~1;
  int hAdd = (kl0 & 8) ? 16 : 0;      // +16 lanes region of A layout
  int hv   = ((kl0 & 16) >> 2) + ((kl0 & 7) >> 1);

  v8f out_acc[12];
#pragma unroll
  for (int i = 0; i < 12; ++i)
    out_acc[i] = (v8f){0.f, 0.f, 0.f, 0.f, 0.f, 0.f, 0.f, 0.f};

  for (int hb = 0; hb < H_DIM; hb += HB) {
    // ---- stage 1: h_chunk = leaky(x @ W1[:, hb..hb+64) + b1), wave -> one 16x16 tile
    {
      v8f hacc = (v8f){0.f, 0.f, 0.f, 0.f, 0.f, 0.f, 0.f, 0.f};
      int hT = (hb >> 4) + wc;                                  // global H n-tile
      const uint4* bp = (const uint4*)(w1t + (((size_t)hT * W1_KT) << 8)) + lane * 2;
      const uint4* ap = x_a4 + (mrow * 24) * 64 + lane * 2;
#pragma unroll
      for (int kt = 0; kt < 24; ++kt) {
        B128 A, B;
        A.q[0] = ap[0]; A.q[1] = ap[1];
        B.q[0] = bp[0]; B.q[1] = bp[1];
        hacc = wmma_bf16(A, B, hacc);
        ap += 64; bp += 64;
      }
      float bias = b1p[hb + wc * 16 + l15];
      unsigned short* hu = (unsigned short*)h_a4;
#pragma unroll
      for (int r = 0; r < 8; ++r) {
        float hvf = hacc[r] + bias;
        hvf = (hvf > 0.f) ? hvf : 0.01f * hvf;                  // leaky_relu 0.01
        int m = r + lhalf * 8;                                  // row within mrow tile
        int dwi = ((mrow * 2 + ktl) << 8) + (m + hAdd) * 8 + hv;
        hu[dwi * 2 + hp] = (unsigned short)bf1(hvf);
      }
    }
    __syncthreads();

    // ---- stage 2: out_acc += h_chunk @ W2[hb..hb+64, :]
#pragma unroll
    for (int nt = 0; nt < 12; ++nt) {
      int nT = wc * 12 + nt;                                    // global D n-tile
      const uint4* bp = (const uint4*)(w2t + ((((size_t)nT * W2_KT) + (hb >> 5)) << 8))
                        + lane * 2;
#pragma unroll
      for (int ks = 0; ks < 2; ++ks) {
        const uint4* ap = h_a4 + (mrow * 2 + ks) * 64 + lane * 2;
        B128 A, B;
        A.q[0] = ap[0]; A.q[1] = ap[1];
        B.q[0] = bp[0]; B.q[1] = bp[1];
        out_acc[nt] = wmma_bf16(A, B, out_acc[nt]);
        bp += 64;
      }
    }
    __syncthreads();
  }

  // ---- epilogue: + b2, gate per token row, scatter to per-(t,k) slot ----
#pragma unroll
  for (int nt = 0; nt < 12; ++nt) {
    int col = wc * 192 + nt * 16 + l15;
    float bias2 = b2p[col];
#pragma unroll
    for (int r = 0; r < 8; ++r) {
      int m = mrow * 16 + r + lhalf * 8;
      int slot = slot_l[m];
      if (slot >= 0) {
        float yv = out_acc[nt][r] + bias2;
        if (isShared) sharedOut[(size_t)slot * D_DIM + col] = yv;
        else          partial[(size_t)slot * D_DIM + col]   = yv * gate_l[m];
      }
    }
  }
}

// ---------------- Combine: softmax(alpha,beta) blend ------------------------
__global__ void moe_combine(const float* __restrict__ sharedOut,
                            const float* __restrict__ partial,
                            const float* __restrict__ alpha,
                            const float* __restrict__ beta,
                            float* __restrict__ out) {
  int idx = blockIdx.x * blockDim.x + threadIdx.x;
  if (idx >= T_NUM * D_DIM) return;
  int t = idx / D_DIM, c = idx - t * D_DIM;
  float a = alpha[0], b = beta[0];
  float mx = fmaxf(a, b);
  float ea = __expf(a - mx), eb = __expf(b - mx);
  float inv = 1.f / (ea + eb);
  float moe = partial[(size_t)(2 * t) * D_DIM + c] +
              partial[(size_t)(2 * t + 1) * D_DIM + c];
  out[idx] = (ea * inv) * sharedOut[idx] + (eb * inv) * moe;
}

extern "C" void kernel_launch(void* const* d_in, const int* in_sizes, int n_in,
                              void* d_out, int out_size, void* d_ws, size_t ws_size,
                              hipStream_t stream) {
  const float* x     = (const float*)d_in[0];
  const float* noise = (const float*)d_in[1];
  const float* Wr    = (const float*)d_in[2];
  const float* br    = (const float*)d_in[3];
  const float* W1    = (const float*)d_in[4];
  const float* b1    = (const float*)d_in[5];
  const float* W2    = (const float*)d_in[6];
  const float* b2    = (const float*)d_in[7];
  const float* Ws1   = (const float*)d_in[8];
  const float* bs1   = (const float*)d_in[9];
  const float* Ws2   = (const float*)d_in[10];
  const float* bs2   = (const float*)d_in[11];
  const float* alpha = (const float*)d_in[12];
  const float* beta  = (const float*)d_in[13];
  float* out = (float*)d_out;

  char* ws = (char*)d_ws;
  int*   cnt       = (int*)ws;            ws += 256;
  int*   listSlot  = (int*)ws;            ws += (size_t)E_NUM * T_NUM * 4;
  float* listGate  = (float*)ws;          ws += (size_t)E_NUM * T_NUM * 4;
  float* partial   = (float*)ws;          ws += (size_t)2 * T_NUM * D_DIM * 4;
  float* sharedOut = (float*)ws;          ws += (size_t)T_NUM * D_DIM * 4;
  unsigned int* W1t  = (unsigned int*)ws; ws += (size_t)E_NUM * W1T_DW * 4;
  unsigned int* W2t  = (unsigned int*)ws; ws += (size_t)E_NUM * W2T_DW * 4;
  unsigned int* Ws1t = (unsigned int*)ws; ws += W1T_DW * 4;
  unsigned int* Ws2t = (unsigned int*)ws; ws += W2T_DW * 4;

  hipMemsetAsync(cnt, 0, E_NUM * sizeof(int), stream);
  moe_router<<<T_NUM / 8, 256, 0, stream>>>(x, noise, Wr, br, cnt, listSlot, listGate);

  // weight packing: fp32 row-major -> bf16 WMMA-B tiles
  {
    dim3 g1((D_DIM / 2) * H_DIM / 256, E_NUM);
    pack_w<<<g1, 256, 0, stream>>>(W1, W1t, D_DIM, H_DIM);
    dim3 g2((H_DIM / 2) * D_DIM / 256, E_NUM);
    pack_w<<<g2, 256, 0, stream>>>(W2, W2t, H_DIM, D_DIM);
    dim3 g3((D_DIM / 2) * H_DIM / 256, 1);
    pack_w<<<g3, 256, 0, stream>>>(Ws1, Ws1t, D_DIM, H_DIM);
    dim3 g4((H_DIM / 2) * D_DIM / 256, 1);
    pack_w<<<g4, 256, 0, stream>>>(Ws2, Ws2t, H_DIM, D_DIM);
  }

  moe_ffn<<<(E_NUM + 1) * (T_NUM / TILE_M), 256, 0, stream>>>(
      x, W1t, b1, W2t, b2, Ws1t, bs1, Ws2t, bs2, cnt, listSlot, listGate,
      partial, sharedOut);

  int total = T_NUM * D_DIM;
  moe_combine<<<(total + 255) / 256, 256, 0, stream>>>(sharedOut, partial, alpha, beta, out);
}
